// RowLSTM_36326833389982
// MI455X (gfx1250) — matrix operand order repaired and checked by reading, not compiled, not run on hardware
//
#include <hip/hip_runtime.h>
#include <hip/hip_bf16.h>
#include <cstdint>
#include <cstddef>

// ---------------------------------------------------------------------------
// RowLSTM for MI455X (gfx1250): f16 WMMA GEMMs (wave32), TDM-staged weight
// tiles with hardware pitch padding, software-pipelined fragment loads.
// ---------------------------------------------------------------------------

typedef _Float16 h16;
typedef _Float16 v8h  __attribute__((ext_vector_type(8)));
typedef _Float16 v16h __attribute__((ext_vector_type(16)));
typedef float    v8f  __attribute__((ext_vector_type(8)));
typedef unsigned int v4u __attribute__((ext_vector_type(4)));
typedef int      v4i  __attribute__((ext_vector_type(4)));
typedef int      v8i  __attribute__((ext_vector_type(8)));

#define B_SZ   32
#define HC     128
#define H_IMG  64
#define W_IMG  64
#define L_NUM  7
#define OC     768          // C*NC = 3*256
#define G4     512          // 4*HC
#define PITCH  136          // LDS pitch in halves: 272B rows -> 16B aligned, 4-dword bank stagger
#define ACHUNK (3 * 128 * 128)   // one (dt,op) prepacked A chunk, halves

#if __has_builtin(__builtin_amdgcn_tensor_load_to_lds) && __has_builtin(__builtin_amdgcn_s_wait_tensorcnt)
#define USE_TDM 1
#else
#define USE_TDM 0
#endif

// ---- fragment loaders (CDNA5 ISA VGPR layouts, wave32) ----------------
// A 16x32 f16: lanes 0-15 row M=lane, halves = K[0..7] then K[16..23];
//              lanes 16-31 same row, halves = K[8..15] then K[24..31].
__device__ __forceinline__ v16h ldA16(const h16* p) {
  v8h lo = *(const v8h*)(p);
  v8h hi = *(const v8h*)(p + 16);
  return __builtin_shufflevector(lo, hi, 0,1,2,3,4,5,6,7,8,9,10,11,12,13,14,15);
}
// B 32x16 f16: lanes 0-15 col N=lane hold K[0..7] then K[16..23]; lanes 16-31 hold K[8..15] then K[24..31].
__device__ __forceinline__ v16h ldB16(const h16* p) {
  v8h lo = *(const v8h*)(p);
  v8h hi = *(const v8h*)(p + 16);
  return __builtin_shufflevector(lo, hi, 0,1,2,3,4,5,6,7,8,9,10,11,12,13,14,15);
}
__device__ __forceinline__ float sigmoidf_(float x) { return 1.f / (1.f + __expf(-x)); }

#if USE_TDM
// TDM: linear 1D tile of nelem f16 -> LDS at lds_off, HW-padded 4 DWORDs every
// 64 DWORDs (global pitch 128 halves -> LDS pitch 136 halves).
__device__ __forceinline__ void tdm_load_padded(const void* gsrc, unsigned lds_off,
                                                unsigned nelem) {
  unsigned long long ga = (unsigned long long)(uintptr_t)gsrc;
  v4u g0;
  g0.x = 1u;                                             // count=1, user descriptor
  g0.y = lds_off;                                        // lds_addr (bytes)
  g0.z = (unsigned)ga;                                   // global_addr[31:0]
  g0.w = (unsigned)((ga >> 32) & 0x01FFFFFFu) | (2u << 30); // [56:32] | type=2
  v8i g1;
  g1[0] = (int)((1u << 16)        // data_size = 2 bytes
              | (1u << 20)        // pad_enable
              | (5u << 22)        // pad_interval: 64 DWORDs
              | (3u << 25));      // pad_amount: 4 DWORDs
  g1[1] = (int)((nelem & 0xFFFFu) << 16);                // tensor_dim0[15:0]
  g1[2] = (int)((nelem >> 16) | (1u << 16));             // tensor_dim0[31:16] | tensor_dim1=1
  g1[3] = (int)((nelem & 0xFFFFu) << 16);                // tile_dim0
  g1[4] = 1;                                             // tile_dim1 = 1
  g1[5] = (int)nelem;                                    // tensor_dim0_stride[31:0]
  g1[6] = 0;
  g1[7] = 0;
  v4i z4 = {0, 0, 0, 0};
#if __clang_major__ >= 23
  v8i z8 = {0, 0, 0, 0, 0, 0, 0, 0};
  __builtin_amdgcn_tensor_load_to_lds(g0, g1, z4, z4, z8, 0);
#else
  __builtin_amdgcn_tensor_load_to_lds(g0, g1, z4, z4, 0);
#endif
}
#endif

// ---------------------------------------------------------------------------
__global__ void zero_ws(uint4* p, int n4) {
  for (int i = blockIdx.x * blockDim.x + threadIdx.x; i < n4; i += gridDim.x * blockDim.x)
    p[i] = make_uint4(0u, 0u, 0u, 0u);
}

// pack Wx/Wh -> f16 [l][dt(4)][op(2)][tap(3)][mL(128)][ic(128)] so each block's
// A tile is one contiguous 96KB chunk (TDM-friendly); W_out -> f16 [o][d].
__global__ void prep_weights(const float* __restrict__ Wx, const float* __restrict__ Wh,
                             const float* __restrict__ Wout,
                             h16* __restrict__ wpack, h16* __restrict__ wouth) {
  const int NW = L_NUM * 4 * 2 * ACHUNK;            // 2,752,512
  const int NT = NW + OC * HC;                      // + 98,304
  for (int idx = blockIdx.x * blockDim.x + threadIdx.x; idx < NT; idx += gridDim.x * blockDim.x) {
    if (idx < NW) {
      int ic = idx & 127;  int t = idx >> 7;
      int mL = t & 127;    t >>= 7;
      int tap = t % 3;     t /= 3;
      int op = t & 1;      t >>= 1;
      int dt = t & 3;      int l = t >> 2;
      int m = (mL >> 5) * 128 + dt * 32 + (mL & 31);   // gate-interleaved global row
      const float* src = op ? Wh : Wx;
      wpack[idx] = (h16)src[(((size_t)l * G4 + m) * HC + ic) * 3 + tap];
    } else {
      int j = idx - NW;
      wouth[j] = (h16)Wout[j];
    }
  }
}

// input projection: feat[b][d][y][w] = f16( sum_c x/255 * W_in[d][c] + b_in[d] )
__global__ void input_proj(const int* __restrict__ x, const float* __restrict__ Win,
                           const float* __restrict__ bin, h16* __restrict__ feat) {
  const int tot = B_SZ * HC * H_IMG * W_IMG;
  for (int idx = blockIdx.x * blockDim.x + threadIdx.x; idx < tot; idx += gridDim.x * blockDim.x) {
    int r = idx & 4095; int t = idx >> 12;
    int d = t & 127;    int b = t >> 7;
    float s = bin[d];
#pragma unroll
    for (int c = 0; c < 3; ++c)
      s += ((float)x[(((size_t)b * 3 + c) << 12) + r]) * (1.f / 255.f) * Win[d * 3 + c];
    feat[idx] = (h16)s;
  }
}

// ---------------------------------------------------------------------------
// One (row, layer) step. grid (4,16): blockIdx.x -> 32-ch gate slice d0,
// blockIdx.y -> 128 cols. 8 waves = 4(M) x 2(N); wave tile 32x64.
// launch_bounds(256,1): occupancy is launch-limited (64 blocks), give the
// allocator full VGPR headroom so nothing spills.
__global__ void __launch_bounds__(256, 1)
lstm_step(const h16* __restrict__ wpack_l,   // [4][2][3][128][128] f16 prepacked
          const float* __restrict__ bx_l,    // [512] f32
          const h16* __restrict__ cur_base, int cur_stride,
          const h16* __restrict__ hold_base, // stride 64
          h16* __restrict__ hnew_base,
          float* __restrict__ c_base,
          h16* __restrict__ outh_base, int write_out) {
  extern __shared__ char smem_raw[];
  h16* As = (h16*)smem_raw;                              // [3][128][PITCH]
  h16* BT = (h16*)(smem_raw + 3 * 128 * PITCH * 2);      // [132][PITCH] col-major act, 2-col zero pad
  float* comb = (float*)smem_raw;                        // reuse As region: [128][132] f32

  const int d0 = blockIdx.x * 32;
  const int n0 = blockIdx.y * 128;
  const int tid = threadIdx.x;
  const int wv = tid >> 5, lane = tid & 31;
  const int lane15 = lane & 15, hiHalf = lane >> 4;
  const int mg = wv >> 1, ng = wv & 1;                   // 4 x 2 wave grid
  const int mRow0 = mg * 32 + lane15;
  const int mRow1 = mRow0 + 16;

  v8f acc0[4], acc1[4];
  const v8f vz = {0.f, 0.f, 0.f, 0.f, 0.f, 0.f, 0.f, 0.f};
#pragma unroll
  for (int i = 0; i < 4; ++i) { acc0[i] = vz; acc1[i] = vz; }

  for (int op = 0; op < 2; ++op) {
    __syncthreads();
    const h16* wsrc = wpack_l + (size_t)(blockIdx.x * 2 + op) * ACHUNK;
    // ---- stage A tile (96KB contiguous) -----------------------------------
#if USE_TDM
    if (wv == 0)
      tdm_load_padded(wsrc, (unsigned)(size_t)((char*)As - smem_raw), ACHUNK);
#else
    for (int c = tid; c < 3072; c += 256) {              // 384 rows * 8 quads
      int row = c >> 3, q = c & 7;
      *(v8h*)(As + row * PITCH + q * 8) = *(const v8h*)(wsrc + (size_t)row * 128 + q * 8);
    }
#endif
    // ---- stage B (activations, transposed, 2-col causal zero pad) ---------
    {
      const h16* ab = (op == 0) ? cur_base : hold_base;
      const int astr = (op == 0) ? cur_stride : W_IMG;
      int bl = tid >> 7, k = tid & 127;
      int bg = (n0 >> 6) + bl;
      const h16* src = ab + (size_t)(bg * HC + k) * astr;
#pragma unroll 8
      for (int w = 0; w < W_IMG; ++w)
        BT[(bl * 66 + w + 2) * PITCH + k] = src[w];
    }
    for (int i = tid; i < 512; i += 256) {               // zero pad columns
      int bb = i >> 8, pc = (i >> 7) & 1, k = i & 127;
      BT[(bb * 66 + pc) * PITCH + k] = (h16)0;
    }
#if USE_TDM
    if (wv == 0) __builtin_amdgcn_s_wait_tensorcnt(0);
#endif
    __syncthreads();

    // ---- pipelined GEMM: 12 k-steps (3 taps x 4 chunks of K=32) -----------
    // A double-buffered across steps, B look-ahead-by-one fragment; each
    // wait tolerates exactly the in-flight prefetch under a running WMMA.
    v16h a0[2], a1[2], bcur, bnxt;
    auto aAddr = [&](int step, int row) -> const h16* {
      int tap = step >> 2, kc = step & 3;
      return As + (tap * 128 + row) * PITCH + kc * 32 + hiHalf * 8;
    };
    auto bAddr = [&](int step, int nt) -> const h16* {
      int tap = step >> 2, kc = step & 3;
      int n = ng * 64 + nt * 16 + lane15;
      int col = (n >> 6) * 66 + (n & 63) + tap;          // shifted (causal) column
      return BT + col * PITCH + kc * 32 + hiHalf * 8;
    };
    a0[0] = ldA16(aAddr(0, mRow0));
    a1[0] = ldA16(aAddr(0, mRow1));
    bcur  = ldB16(bAddr(0, 0));
#pragma unroll
    for (int step = 0; step < 12; ++step) {
      const int buf = step & 1;
#pragma unroll
      for (int nt = 0; nt < 4; ++nt) {
        if (nt < 3) {
          bnxt = ldB16(bAddr(step, nt + 1));
        } else if (step < 11) {
          a0[buf ^ 1] = ldA16(aAddr(step + 1, mRow0));
          a1[buf ^ 1] = ldA16(aAddr(step + 1, mRow1));
          bnxt = ldB16(bAddr(step + 1, 0));
        }
        acc0[nt] = __builtin_amdgcn_wmma_f32_16x16x32_f16(
            false, a0[buf], false, bcur, (short)0, acc0[nt], false, false);
        acc1[nt] = __builtin_amdgcn_wmma_f32_16x16x32_f16(
            false, a1[buf], false, bcur, (short)0, acc1[nt], false, false);
        bcur = bnxt;
      }
    }
  }
  __syncthreads();
  // ---- spill comb tile to LDS (D layout: VGPR r -> M = r / r+8) -----------
#pragma unroll
  for (int nt = 0; nt < 4; ++nt) {
    int n = ng * 64 + nt * 16 + lane15;
#pragma unroll
    for (int r = 0; r < 8; ++r) {
      comb[(mg * 32 +      hiHalf * 8 + r) * 132 + n] = acc0[nt][r];
      comb[(mg * 32 + 16 + hiHalf * 8 + r) * 132 + n] = acc1[nt][r];
    }
  }
  __syncthreads();
  // ---- LSTM gates for 32 channels x 128 columns ---------------------------
  for (int i = 0; i < 16; ++i) {
    int idx = tid + 256 * i;                             // 4096 elems
    int j = idx >> 7, n = idx & 127;
    float iv = comb[(     j) * 132 + n] + bx_l[0 * HC + d0 + j];
    float fv = comb[(32 + j) * 132 + n] + bx_l[1 * HC + d0 + j];
    float ov = comb[(64 + j) * 132 + n] + bx_l[2 * HC + d0 + j];
    float gv = comb[(96 + j) * 132 + n] + bx_l[3 * HC + d0 + j];
    iv = sigmoidf_(iv); fv = sigmoidf_(fv); ov = sigmoidf_(ov); gv = tanhf(gv);
    int nG = n0 + n, bg = nG >> 6, w = nG & 63, dg = d0 + j;
    size_t ci = ((size_t)bg * HC + dg) * W_IMG + w;
    float cn = fv * c_base[ci] + iv * gv;
    c_base[ci] = cn;
    float hn = ov * tanhf(cn);
    hnew_base[ci] = (h16)hn;
    if (write_out) outh_base[((size_t)bg * HC + dg) * 4096 + w] = (h16)hn;
  }
}

// ---------------------------------------------------------------------------
// Output projection: logits[b][o][r] = sum_d Wout[o][d]*h6[b][d][r] + bout[o]
__global__ void __launch_bounds__(256, 1)
out_proj(const h16* __restrict__ wouth, const float* __restrict__ bout,
         const h16* __restrict__ outh, float* __restrict__ out) {
  extern __shared__ char smem_raw[];
  h16* As = (h16*)smem_raw;                            // [128][PITCH]
  h16* BT = (h16*)(smem_raw + 128 * PITCH * 2);        // [128][PITCH]
  const int m0 = blockIdx.x * 128;
  const int nFlat = blockIdx.y * 128;
  const int b = nFlat >> 12, r0 = nFlat & 4095;
  const int tid = threadIdx.x;

  for (int c = tid; c < 1024; c += 256) {              // 128 rows * 8 quads
    int row = c >> 3, q = c & 7;
    *(v8h*)(As + row * PITCH + q * 8) = *(const v8h*)(wouth + (size_t)(m0 + row) * HC + q * 8);
  }
  {
    int d = tid >> 1, seg = tid & 1;
    const h16* src = outh + ((size_t)b * HC + d) * 4096 + r0 + seg * 64;
    __builtin_prefetch(src, 0, 1);
#pragma unroll 8
    for (int w = 0; w < 64; ++w)
      BT[(seg * 64 + w) * PITCH + d] = src[w];
  }
  __syncthreads();

  const int wv = tid >> 5, lane = tid & 31;
  const int lane15 = lane & 15, hiHalf = lane >> 4;
  const int mg = wv >> 1, ng = wv & 1;
  const int mRow0 = mg * 32 + lane15, mRow1 = mRow0 + 16;
  v8f acc0[4], acc1[4];
  const v8f vz = {0.f, 0.f, 0.f, 0.f, 0.f, 0.f, 0.f, 0.f};
#pragma unroll
  for (int i = 0; i < 4; ++i) { acc0[i] = vz; acc1[i] = vz; }

  v16h a0[2], a1[2], bcur, bnxt;
  auto bAddr = [&](int kc, int nt) -> const h16* {
    return BT + (ng * 64 + nt * 16 + lane15) * PITCH + kc * 32 + hiHalf * 8;
  };
  a0[0] = ldA16(As + mRow0 * PITCH + hiHalf * 8);
  a1[0] = ldA16(As + mRow1 * PITCH + hiHalf * 8);
  bcur  = ldB16(bAddr(0, 0));
#pragma unroll
  for (int kc = 0; kc < 4; ++kc) {
    const int buf = kc & 1;
#pragma unroll
    for (int nt = 0; nt < 4; ++nt) {
      if (nt < 3) {
        bnxt = ldB16(bAddr(kc, nt + 1));
      } else if (kc < 3) {
        a0[buf ^ 1] = ldA16(As + mRow0 * PITCH + (kc + 1) * 32 + hiHalf * 8);
        a1[buf ^ 1] = ldA16(As + mRow1 * PITCH + (kc + 1) * 32 + hiHalf * 8);
        bnxt = ldB16(bAddr(kc + 1, 0));
      }
      acc0[nt] = __builtin_amdgcn_wmma_f32_16x16x32_f16(
          false, a0[buf], false, bcur, (short)0, acc0[nt], false, false);
      acc1[nt] = __builtin_amdgcn_wmma_f32_16x16x32_f16(
          false, a1[buf], false, bcur, (short)0, acc1[nt], false, false);
      bcur = bnxt;
    }
  }
#pragma unroll
  for (int nt = 0; nt < 4; ++nt) {
    int n = ng * 64 + nt * 16 + lane15;
#pragma unroll
    for (int r = 0; r < 8; ++r) {
      int ma = m0 + mg * 32 + hiHalf * 8 + r;
      int mb = ma + 16;
      out[((size_t)b * OC + ma) * 4096 + r0 + n] = acc0[nt][r] + bout[ma];
      out[((size_t)b * OC + mb) * 4096 + r0 + n] = acc1[nt][r] + bout[mb];
    }
  }
}

// ---------------------------------------------------------------------------
extern "C" void kernel_launch(void* const* d_in, const int* in_sizes, int n_in,
                              void* d_out, int out_size, void* d_ws, size_t ws_size,
                              hipStream_t stream) {
  (void)in_sizes; (void)n_in; (void)out_size; (void)ws_size;
  const int*   x     = (const int*)  d_in[0];
  const float* W_in  = (const float*)d_in[1];
  const float* b_in  = (const float*)d_in[2];
  const float* Wx    = (const float*)d_in[3];
  const float* bx    = (const float*)d_in[4];
  const float* Wh    = (const float*)d_in[5];
  const float* W_out = (const float*)d_in[6];
  const float* b_out = (const float*)d_in[7];
  float* out = (float*)d_out;

  // workspace carve (~87.5 MB total) -- all offsets 256B aligned
  char* ws = (char*)d_ws;
  h16* wpack = (h16*)ws;   ws += (size_t)L_NUM * 4 * 2 * ACHUNK * 2;         // 5,505,024
  h16* wouth = (h16*)ws;   ws += (size_t)OC * HC * 2;                        //   196,608
  h16* feat  = (h16*)ws;   ws += (size_t)B_SZ * HC * H_IMG * W_IMG * 2;      // 33,554,432
  h16* hstate= (h16*)ws;   ws += (size_t)2 * L_NUM * B_SZ * HC * W_IMG * 2;  //  7,340,032
  float* cst = (float*)ws; ws += (size_t)L_NUM * B_SZ * HC * W_IMG * 4;      //  7,340,032
  h16* outh  = (h16*)ws;                                                     // 33,554,432

  const size_t lds_lstm = (size_t)(3 * 128 * PITCH + 132 * PITCH) * 2;   // 140,352 B (<320KB WGP LDS)
  const size_t lds_out  = (size_t)(2 * 128 * PITCH) * 2;                 //  69,632 B
  (void)hipFuncSetAttribute((const void*)lstm_step,
                            hipFuncAttributeMaxDynamicSharedMemorySize, (int)lds_lstm);
  (void)hipFuncSetAttribute((const void*)out_proj,
                            hipFuncAttributeMaxDynamicSharedMemorySize, (int)lds_out);

  // 1) zero h/c recurrent state (contiguous region)
  {
    int n4 = (int)(((size_t)2 * L_NUM * B_SZ * HC * W_IMG * 2 +
                    (size_t)L_NUM * B_SZ * HC * W_IMG * 4) / 16);
    zero_ws<<<3584, 256, 0, stream>>>((uint4*)hstate, n4);
  }
  // 2) pack weights to f16 (contiguous per-block A chunks)
  prep_weights<<<2048, 256, 0, stream>>>(Wx, Wh, W_out, wpack, wouth);
  // 3) input projection
  input_proj<<<4096, 256, 0, stream>>>(x, W_in, b_in, feat);

  // 4) row scan: 64 rows x 7 layers, parity double-buffered h-state
  const size_t lstride = (size_t)B_SZ * HC * W_IMG;   // per-layer state elems
  for (int y = 0; y < H_IMG; ++y) {
    int p = y & 1;
    h16* hOld = hstate + (size_t)p * L_NUM * lstride;
    h16* hNew = hstate + (size_t)(1 - p) * L_NUM * lstride;
    for (int l = 0; l < L_NUM; ++l) {
      const h16* cur; int curStride;
      if (l == 0) { cur = feat + (size_t)y * W_IMG; curStride = H_IMG * W_IMG; }
      else        { cur = hNew + (size_t)(l - 1) * lstride; curStride = W_IMG; }
      lstm_step<<<dim3(4, 16), 256, lds_lstm, stream>>>(
          wpack + (size_t)l * 4 * 2 * ACHUNK,
          bx + l * G4,
          cur, curStride,
          hOld + (size_t)l * lstride,
          hNew + (size_t)l * lstride,
          cst  + (size_t)l * lstride,
          outh + (size_t)y * W_IMG, (l == L_NUM - 1) ? 1 : 0);
    }
  }
  // 5) output projection
  out_proj<<<dim3(6, 1024), 256, lds_out, stream>>>(wouth, b_out, outh, out);
}
